// DeepseekV2AttentionMLA_44100724195695
// MI455X (gfx1250) — matrix operand order repaired and checked by reading, not compiled, not run on hardware
//
#include <hip/hip_runtime.h>
#include <hip/hip_bf16.h>

// DeepSeek-V2 MLA attention for MI455X (gfx1250, wave32, WMMA).
// Pipeline: prep (f32->bf16 + transposes) -> q_full WMMA GEMM -> fused
// flash-MLA (QK^T WMMA, online softmax, PV WMMA, LSE merge, w_vc WMMA).
// Workspace use: ~41 MB (kv bf16 9.4M + kvT 8.4M + wkcT 2.1M + wvcT 2.1M + qfull 18.9M).
// CDNA5 paths: v_wmma_f32_16x16x32_bf16 throughout; async global->LDS copy of
// the q tile (ASYNCcnt + s_wait_asynccnt); global_prefetch for next-step kv slabs.

#define T_DIM 1024
#define H_DIM 16
#define S_DIM 8192
#define NOPE 128
#define ROPE 64
#define LORA 512
#define VDIM 128
#define DTOT 576   // LORA + ROPE
#define QDIM 192   // NOPE + ROPE
#define SCALING 0.07216878364870322f   // (NOPE+ROPE)^-0.5

typedef __attribute__((ext_vector_type(16))) __bf16 v16bf;
typedef __attribute__((ext_vector_type(8)))  float  v8f;

#if __has_builtin(__builtin_amdgcn_global_load_async_to_lds_b128) && \
    __has_builtin(__builtin_amdgcn_s_wait_asynccnt)
#define HAVE_ASYNC_LDS 1
// builtin signature: (v4i addrspace(1)*, v4i addrspace(3)*, imm offset, imm cpol)
typedef int async_v4i __attribute__((vector_size(16)));
typedef __attribute__((address_space(1))) async_v4i* g_v4i_ptr;
typedef __attribute__((address_space(3))) async_v4i* l_v4i_ptr;
#else
#define HAVE_ASYNC_LDS 0
#endif

// Map a K-dim index d (within a 576-dim row) to its storage slot: identity, so a
// contiguous 16-element load per lane yields lane-half g holding K = g*16..g*16+15
// per 32-slice (matching the contiguous B loads).
__device__ __forceinline__ int slot_of_dim(int d) {
  return d;
}

// ---------------------------------------------------------------- prep: kv
__global__ void prep_kv_kernel(const float* __restrict__ kv,
                               __bf16* __restrict__ kvb,
                               __bf16* __restrict__ kvT) {
  int i = blockIdx.x * blockDim.x + threadIdx.x;
  if (i >= S_DIM * DTOT) return;
  int s = i / DTOT, d = i % DTOT;
  __bf16 v = (__bf16)kv[i];
  kvb[i] = v;                                     // row-major (s, d) for QK^T B-frags
  if (d < LORA) kvT[(size_t)d * S_DIM + s] = v;   // (l, s) for PV B-frags
}

// ------------------------------------------------------------- prep: weights
__global__ void prep_w_kernel(const float* __restrict__ wkc,
                              const float* __restrict__ wvc,
                              __bf16* __restrict__ wkcT,
                              __bf16* __restrict__ wvcT) {
  int i = blockIdx.x * blockDim.x + threadIdx.x;
  if (i >= H_DIM * NOPE * LORA) return;
  {
    int h = i / (NOPE * LORA), r = i % (NOPE * LORA);
    int n = r / LORA, l = r % LORA;
    wkcT[((size_t)h * LORA + l) * NOPE + n] = (__bf16)wkc[i];   // (h, l, nope)
  }
  {
    int h = i / (LORA * VDIM), r = i % (LORA * VDIM);
    int l = r / VDIM, v = r % VDIM;
    wvcT[((size_t)h * VDIM + v) * LORA + l] = (__bf16)wvc[i];   // (h, v, lora)
  }
}

// -------------------------------------------------- q_full = [q_nope@w_kc, q_pe]
__global__ void __launch_bounds__(128)
qfull_kernel(const float* __restrict__ q, const __bf16* __restrict__ wkcT,
             __bf16* __restrict__ qf) {
  const int t0 = blockIdx.x * 16;
  const int h  = blockIdx.y;
  const int tid  = threadIdx.x;
  const int lane = tid & 31, wv = tid >> 5;
  const int lm = lane & 15, lh = lane >> 4;

  v8f c[8];
  #pragma unroll
  for (int nt = 0; nt < 8; ++nt) {
    #pragma unroll
    for (int r = 0; r < 8; ++r) c[nt][r] = 0.f;
  }

  const float* qrow = q + ((size_t)(t0 + lm) * H_DIM + h) * QDIM;
  #pragma unroll
  for (int j = 0; j < 4; ++j) {               // K slices of 32 over NOPE=128
    v16bf a;
    #pragma unroll
    for (int i = 0; i < 16; ++i) {
      int k = j * 32 + lh * 16 + i;
      a[i] = (__bf16)qrow[k];
    }
    #pragma unroll
    for (int nt = 0; nt < 8; ++nt) {          // cols wv*128 .. wv*128+127
      int col = wv * 128 + nt * 16 + lm;
      const __bf16* bp = wkcT + ((size_t)h * LORA + col) * NOPE + j * 32 + lh * 16;
      v16bf b = *(const v16bf*)bp;
      c[nt] = __builtin_amdgcn_wmma_f32_16x16x32_bf16(false, a, false, b,
                                                      (short)0, c[nt], false, false);
    }
  }
  // store q_lora, scaled, into identity (natural dim-order) slots
  #pragma unroll
  for (int nt = 0; nt < 8; ++nt) {
    #pragma unroll
    for (int r = 0; r < 8; ++r) {
      int row = r + 8 * lh;
      int d   = wv * 128 + nt * 16 + lm;
      qf[((size_t)(t0 + row) * H_DIM + h) * DTOT + slot_of_dim(d)] =
          (__bf16)(c[nt][r] * SCALING);
    }
  }
  // q_pe part (dims 512..575), cooperative
  int prow = tid >> 3;          // 0..15
  int peb  = (tid & 7) * 8;     // 0..56
  const float* qsrc = q + ((size_t)(t0 + prow) * H_DIM + h) * QDIM + NOPE;
  __bf16* qdst = qf + ((size_t)(t0 + prow) * H_DIM + h) * DTOT;
  #pragma unroll
  for (int i = 0; i < 8; ++i) {
    int pe = peb + i;
    qdst[slot_of_dim(LORA + pe)] = (__bf16)(qsrc[pe] * SCALING);
  }
}

// ------------------------------------------------------- fused flash-MLA core
__global__ void __launch_bounds__(128)
mla_attn_kernel(const __bf16* __restrict__ qf, const __bf16* __restrict__ kvb,
                const __bf16* __restrict__ kvT, const __bf16* __restrict__ wvcT,
                float* __restrict__ out) {
  __shared__ __bf16 lds_q[16 * DTOT];        // 18 KB  shared q tile
  __shared__ __bf16 lds_p[4][16 * 32];       // 4 KB   per-wave P staging (bf16, A-frag order)
  __shared__ float  lds_o[16 * LORA];        // 32 KB  merged attention output
  __shared__ float  lds_m[4][16];
  __shared__ float  lds_l[4][16];

  const int t0 = blockIdx.x * 16;
  const int h  = blockIdx.y;
  const int tid  = threadIdx.x;
  const int lane = tid & 31, wv = tid >> 5;
  const int lm = lane & 15, lh = lane >> 4;

  // load the 16x576 bf16 q tile into LDS
#if HAVE_ASYNC_LDS
  // async global->LDS (tracked by ASYNCcnt, no VGPR staging):
  // 1152 16-byte chunks, 9 per thread.
  for (int c = tid; c < 16 * 72; c += 128) {
    int m = c / 72, dw = c % 72;                       // 72 x 16B per 576-elem row
    const __bf16* g = qf + ((size_t)(t0 + m) * H_DIM + h) * DTOT + dw * 8;
    __bf16* l = lds_q + m * DTOT + dw * 8;
    __builtin_amdgcn_global_load_async_to_lds_b128(
        (g_v4i_ptr)(g), (l_v4i_ptr)(l), 0, 0);
  }
  __builtin_amdgcn_s_wait_asynccnt(0);
#else
  {
    const unsigned int* src = (const unsigned int*)qf;
    unsigned int* dst = (unsigned int*)lds_q;
    for (int i = tid; i < 16 * 288; i += 128) {
      int m = i / 288, dw = i % 288;
      dst[m * 288 + dw] = src[((size_t)(t0 + m) * H_DIM + h) * 288 + dw];
    }
  }
#endif
  __syncthreads();

  v8f acc[32];
  #pragma unroll
  for (int nt = 0; nt < 32; ++nt) {
    #pragma unroll
    for (int r = 0; r < 8; ++r) acc[nt][r] = 0.f;
  }
  float mrow[8], lrow[8];
  #pragma unroll
  for (int r = 0; r < 8; ++r) { mrow[r] = -INFINITY; lrow[r] = 0.f; }

  const int chunk0 = wv * (S_DIM / 4);       // each wave owns one S-chunk (as in reference)
  // P-store slots (contiguous positions for keys lm and 16+lm)
  const int p0pos = lm;
  const int p1pos = 16 + lm;

  #pragma unroll 1
  for (int step = 0; step < (S_DIM / 4) / 32; ++step) {
    const int sbase = chunk0 + step * 32;

    const __bf16* brow0 = kvb + (size_t)(sbase + lm) * DTOT + lh * 16;
    const __bf16* brow1 = brow0 + (size_t)16 * DTOT;
    const __bf16* arow  = lds_q + lm * DTOT + lh * 16;
    const __bf16* vbase = kvT + (size_t)lm * S_DIM + sbase + lh * 16;

    // prefetch next step's kv slabs (L2-resident; hides L2 latency behind WMMAs)
    __builtin_prefetch(brow0 + (size_t)32 * DTOT, 0, 0);
    __builtin_prefetch(brow1 + (size_t)32 * DTOT, 0, 0);
    __builtin_prefetch(vbase + 32, 0, 0);

    // ---- scores: 16 rows x 32 keys, K = 576
    v8f c0 = {}, c1 = {};
    #pragma unroll
    for (int j = 0; j < 18; ++j) {
      v16bf a  = *(const v16bf*)(arow  + j * 32);
      v16bf b0 = *(const v16bf*)(brow0 + j * 32);
      v16bf b1 = *(const v16bf*)(brow1 + j * 32);
      c0 = __builtin_amdgcn_wmma_f32_16x16x32_bf16(false, a, false, b0, (short)0, c0, false, false);
      c1 = __builtin_amdgcn_wmma_f32_16x16x32_bf16(false, a, false, b1, (short)0, c1, false, false);
    }

    // ---- online softmax: rows live across 16-lane halves of the C layout
    float sc[8];
    #pragma unroll
    for (int r = 0; r < 8; ++r) {
      float v = fmaxf(c0[r], c1[r]);
      v = fmaxf(v, __shfl_xor(v, 1, 32));
      v = fmaxf(v, __shfl_xor(v, 2, 32));
      v = fmaxf(v, __shfl_xor(v, 4, 32));
      v = fmaxf(v, __shfl_xor(v, 8, 32));
      float mn = fmaxf(mrow[r], v);
      sc[r] = __expf(mrow[r] - mn);
      float e0 = __expf(c0[r] - mn);
      float e1 = __expf(c1[r] - mn);
      float sum = e0 + e1;
      sum += __shfl_xor(sum, 1, 32);
      sum += __shfl_xor(sum, 2, 32);
      sum += __shfl_xor(sum, 4, 32);
      sum += __shfl_xor(sum, 8, 32);
      lrow[r] = lrow[r] * sc[r] + sum;
      mrow[r] = mn;
      int row = r + 8 * lh;
      lds_p[wv][row * 32 + p0pos] = (__bf16)e0;
      lds_p[wv][row * 32 + p1pos] = (__bf16)e1;
    }
    // rescale running accumulators
    #pragma unroll
    for (int nt = 0; nt < 32; ++nt) {
      #pragma unroll
      for (int r = 0; r < 8; ++r) acc[nt][r] *= sc[r];
    }

    // ---- P @ V over 512 lora cols (per-wave LDS buffer; same-wave DS ops are in order)
    v16bf pa = *(const v16bf*)(&lds_p[wv][lm * 32 + lh * 16]);
    #pragma unroll
    for (int nt = 0; nt < 32; ++nt) {
      v16bf b = *(const v16bf*)(vbase + (size_t)(nt * 16) * S_DIM);
      acc[nt] = __builtin_amdgcn_wmma_f32_16x16x32_bf16(false, pa, false, b,
                                                        (short)0, acc[nt], false, false);
    }
  }

  // ---- LSE merge across the 4 wave-chunks (reference's logaddexp scan)
  if (lm == 0) {
    #pragma unroll
    for (int r = 0; r < 8; ++r) {
      int row = r + 8 * lh;
      lds_m[wv][row] = mrow[r];
      lds_l[wv][row] = lrow[r];
    }
  }
  __syncthreads();
  float wsc[8];
  #pragma unroll
  for (int r = 0; r < 8; ++r) {
    int row = r + 8 * lh;
    float M = fmaxf(fmaxf(lds_m[0][row], lds_m[1][row]),
                    fmaxf(lds_m[2][row], lds_m[3][row]));
    float L = 0.f;
    #pragma unroll
    for (int w = 0; w < 4; ++w) L += lds_l[w][row] * __expf(lds_m[w][row] - M);
    wsc[r] = __expf(mrow[r] - M) / L;
  }
  #pragma unroll
  for (int nt = 0; nt < 32; ++nt) {
    #pragma unroll
    for (int r = 0; r < 8; ++r) acc[nt][r] *= wsc[r];
  }
  // serialized accumulate into shared output (deterministic)
  for (int w = 0; w < 4; ++w) {
    if (wv == w) {
      #pragma unroll
      for (int nt = 0; nt < 32; ++nt) {
        #pragma unroll
        for (int r = 0; r < 8; ++r) {
          int row = r + 8 * lh, col = nt * 16 + lm;
          if (w == 0) lds_o[row * LORA + col]  = acc[nt][r];
          else        lds_o[row * LORA + col] += acc[nt][r];
        }
      }
    }
    __syncthreads();
  }

  // ---- final projection: attn (16x512) @ w_vc[h] (512x128); wave wv -> cols wv*32..+31
  const int n0 = wv * 32;
  v8f o0 = {}, o1 = {};
  const __bf16* wvp0 = wvcT + ((size_t)h * VDIM + n0 + lm) * LORA + lh * 16;
  const __bf16* wvp1 = wvp0 + (size_t)16 * LORA;
  #pragma unroll
  for (int ks = 0; ks < 16; ++ks) {
    v16bf a;
    #pragma unroll
    for (int i = 0; i < 16; ++i) {
      int k = ks * 32 + lh * 16 + i;
      a[i] = (__bf16)lds_o[lm * LORA + k];
    }
    v16bf b0 = *(const v16bf*)(wvp0 + ks * 32);
    v16bf b1 = *(const v16bf*)(wvp1 + ks * 32);
    o0 = __builtin_amdgcn_wmma_f32_16x16x32_bf16(false, a, false, b0, (short)0, o0, false, false);
    o1 = __builtin_amdgcn_wmma_f32_16x16x32_bf16(false, a, false, b1, (short)0, o1, false, false);
  }
  #pragma unroll
  for (int r = 0; r < 8; ++r) {
    int row = r + 8 * lh, col = n0 + lm;
    out[(size_t)(t0 + row) * (H_DIM * VDIM) + h * VDIM + col]      = o0[r];
    out[(size_t)(t0 + row) * (H_DIM * VDIM) + h * VDIM + col + 16] = o1[r];
  }
}

// --------------------------------------------------------------------- launch
extern "C" void kernel_launch(void* const* d_in, const int* in_sizes, int n_in,
                              void* d_out, int out_size, void* d_ws, size_t ws_size,
                              hipStream_t stream) {
  const float* q   = (const float*)d_in[0];
  const float* kv  = (const float*)d_in[1];
  const float* wkc = (const float*)d_in[2];
  const float* wvc = (const float*)d_in[3];
  float* out = (float*)d_out;

  char* ws = (char*)d_ws;
  size_t off = 0;
  auto carve = [&](size_t bytes) -> char* {
    char* p = ws + off;
    off = (off + bytes + 255) & ~(size_t)255;
    return p;
  };
  __bf16* kvb  = (__bf16*)carve((size_t)S_DIM * DTOT * 2);
  __bf16* kvT  = (__bf16*)carve((size_t)LORA * S_DIM * 2);
  __bf16* wkcT = (__bf16*)carve((size_t)H_DIM * LORA * NOPE * 2);
  __bf16* wvcT = (__bf16*)carve((size_t)H_DIM * VDIM * LORA * 2);
  __bf16* qf   = (__bf16*)carve((size_t)T_DIM * H_DIM * DTOT * 2);
  (void)in_sizes; (void)n_in; (void)out_size; (void)ws_size;

  {
    int total = S_DIM * DTOT;
    prep_kv_kernel<<<(total + 255) / 256, 256, 0, stream>>>(kv, kvb, kvT);
  }
  {
    int total = H_DIM * NOPE * LORA;
    prep_w_kernel<<<(total + 255) / 256, 256, 0, stream>>>(wkc, wvc, wkcT, wvcT);
  }
  qfull_kernel<<<dim3(T_DIM / 16, H_DIM), 128, 0, stream>>>(q, wkcT, qf);
  mla_attn_kernel<<<dim3(T_DIM / 16, H_DIM), 128, 0, stream>>>(qf, kvb, kvT, wvcT, out);
}
